// SudokuRRN_42614665511566
// MI455X (gfx1250) — matrix operand order repaired and checked by reading, not compiled, not run on hardware
//
#include <hip/hip_runtime.h>
#include <hip/hip_bf16.h>

typedef unsigned short u16;
typedef unsigned int u32;

typedef __attribute__((ext_vector_type(16))) __bf16 v16bf;
typedef __attribute__((ext_vector_type(8)))  float  v8f;

#define BATCH   128
#define CELLS   81
#define NN      (BATCH * CELLS)   // 10368 nodes
#define HD      96
#define TSTEPS  16
#define DEG     20

// ---------------- bf16 helpers ----------------
__device__ __forceinline__ u16 f2bf(float f) {
  union { float f; u32 u; } x; x.f = f;
  u32 r = x.u + 0x7FFFu + ((x.u >> 16) & 1u);
  return (u16)(r >> 16);
}

union Frag { v16bf v; uint4 q[2]; };

// A fragment (16x32 bf16): row m = lane%16; lanes<16 hold K {kb..kb+7, kb+16..kb+23},
// lanes>=16 hold K {kb+8.., kb+24..} per ISA 16-bit A layout.
__device__ __forceinline__ Frag load_a_bf(const u16* row, int kbase, int lane) {
  const int off = (lane & 16) ? 8 : 0;
  Frag a;
  a.q[0] = *reinterpret_cast<const uint4*>(row + kbase + off);
  a.q[1] = *reinterpret_cast<const uint4*>(row + kbase + off + 16);
  return a;
}

// B fragment (32x16 bf16): lane n = lane%16; lanes<16 hold K {0..7, 16..23}, lanes>=16 hold K {8..15, 24..31}.
// With row-major W[out][K], B[k][n] = W[n][k] -> two 16-byte chunks.
__device__ __forceinline__ Frag load_b_bf(const u16* W, int ld, int n, int kt, int lane) {
  const u16* p = W + n * ld + kt * 32 + ((lane & 16) ? 8 : 0);
  Frag b;
  b.q[0] = reinterpret_cast<const uint4*>(p)[0];
  b.q[1] = reinterpret_cast<const uint4*>(p + 16)[0];
  return b;
}

// A fragment built from an f32 row (LDS message accumulator) with on-the-fly bf16 convert
__device__ __forceinline__ Frag load_a_f32(const float* row, int kbase, int lane) {
  const int off = (lane & 16) ? 8 : 0;
  const float* lo = row + kbase + off;
  const float* hi = lo + 16;
  u32 w[8];
#pragma unroll
  for (int i = 0; i < 4; ++i)
    w[i] = (u32)f2bf(lo[2 * i]) | ((u32)f2bf(lo[2 * i + 1]) << 16);
#pragma unroll
  for (int i = 0; i < 4; ++i)
    w[4 + i] = (u32)f2bf(hi[2 * i]) | ((u32)f2bf(hi[2 * i + 1]) << 16);
  Frag a;
  a.q[0] = make_uint4(w[0], w[1], w[2], w[3]);
  a.q[1] = make_uint4(w[4], w[5], w[6], w[7]);
  return a;
}

__device__ __forceinline__ v8f wmma_bf16(const Frag& a, const Frag& b, v8f c) {
  return __builtin_amdgcn_wmma_f32_16x16x32_bf16(false, a.v, false, b.v, (short)0, c,
                                                 false, false);
}

__device__ __forceinline__ float sigm(float x) { return 1.f / (1.f + __expf(-x)); }

// ---------------- ws region sizes (elements) ----------------
#define SZ_INW0P (96 * 64)
#define SZ_INW   (3 * 96 * 96)
#define SZ_MW0   (96 * 192)
#define SZ_MW    (3 * 96 * 96)
#define SZ_WIH   (384 * 192)
#define SZ_WHH   (384 * 96)
#define SZ_OUTWP (16 * 96)
#define SZ_DE    (10 * 16)
#define SZ_RE    (9 * 16)
#define SZ_CE    (9 * 16)
#define SZ_ACT   (NN * HD)

// =====================================================================
// prep: convert all weights f32 -> bf16 (with zero padding), zero rc + loss
// =====================================================================
__global__ __launch_bounds__(256) void prep_kernel(
    const float* __restrict__ inW0, const float* __restrict__ inW,
    const float* __restrict__ mW0, const float* __restrict__ mW,
    const float* __restrict__ wih, const float* __restrict__ whh,
    const float* __restrict__ outW, const float* __restrict__ de,
    const float* __restrict__ re, const float* __restrict__ ce,
    u16* __restrict__ dInW0p, u16* __restrict__ dInW, u16* __restrict__ dMW0,
    u16* __restrict__ dMW, u16* __restrict__ dWih, u16* __restrict__ dWhh,
    u16* __restrict__ dOutWp, u16* __restrict__ dDe, u16* __restrict__ dRe,
    u16* __restrict__ dCe, float* __restrict__ rcb, float* __restrict__ lossAcc) {
  long i = (long)blockIdx.x * 256 + threadIdx.x;
  if (i < SZ_INW0P) { int n = i / 64, k = i % 64;
    dInW0p[i] = (k < 48) ? f2bf(inW0[n * 48 + k]) : (u16)0; return; }
  i -= SZ_INW0P;
  if (i < SZ_INW)  { dInW[i] = f2bf(inW[i]); return; }  i -= SZ_INW;
  if (i < SZ_MW0)  { dMW0[i] = f2bf(mW0[i]); return; }  i -= SZ_MW0;
  if (i < SZ_MW)   { dMW[i]  = f2bf(mW[i]);  return; }  i -= SZ_MW;
  if (i < SZ_WIH)  { dWih[i] = f2bf(wih[i]); return; }  i -= SZ_WIH;
  if (i < SZ_WHH)  { dWhh[i] = f2bf(whh[i]); return; }  i -= SZ_WHH;
  if (i < SZ_OUTWP){ int n = i / 96, k = i % 96;
    dOutWp[i] = (n < 10) ? f2bf(outW[n * 96 + k]) : (u16)0; return; }
  i -= SZ_OUTWP;
  if (i < SZ_DE) { dDe[i] = f2bf(de[i]); return; }  i -= SZ_DE;
  if (i < SZ_RE) { dRe[i] = f2bf(re[i]); return; }  i -= SZ_RE;
  if (i < SZ_CE) { dCe[i] = f2bf(ce[i]); return; }  i -= SZ_CE;
  if (i < SZ_ACT){ rcb[i] = 0.f; return; }          i -= SZ_ACT;
  if (i == 0) lossAcc[0] = 0.f;
}
#define PREP_TOTAL (SZ_INW0P + SZ_INW + SZ_MW0 + SZ_MW + SZ_WIH + SZ_WHH + \
                    SZ_OUTWP + SZ_DE + SZ_RE + SZ_CE + SZ_ACT + 1)

// =====================================================================
// input MLP: x = mlp4(cat(d_emb[q], r_emb[row], c_emb[col])) -> xbf (bf16)
// 8 waves x 16 nodes per workgroup (128 nodes / block)
// =====================================================================
__global__ __launch_bounds__(256) void input_kernel(
    const int* __restrict__ q, const int* __restrict__ rowi,
    const int* __restrict__ coli, const u16* __restrict__ eD,
    const u16* __restrict__ eR, const u16* __restrict__ eC,
    const u16* __restrict__ wIn0p, const u16* __restrict__ wIn,
    const float* __restrict__ in_b, u16* __restrict__ xbf) {
  __shared__ u16 sW0p[SZ_INW0P];
  __shared__ u16 sW123[SZ_INW];
  __shared__ u16 sEmb[8][16 * 64];
  __shared__ u16 sInterm[8][2][16 * 96];

  const int tid = threadIdx.x, w = tid >> 5, lane = tid & 31;
  // cooperative weight staging (uint4 = 8 bf16)
  {
    const uint4* s = (const uint4*)wIn0p; uint4* d = (uint4*)sW0p;
    for (int i = tid; i < SZ_INW0P / 8; i += 256) d[i] = s[i];
    const uint4* s1 = (const uint4*)wIn; uint4* d1 = (uint4*)sW123;
    for (int i = tid; i < SZ_INW / 8; i += 256) d1[i] = s1[i];
  }
  const int nb = blockIdx.x * 128 + w * 16;
  // build 16x64 zero-padded embedding tile (K=48 padded to 64)
  {
    const int m = lane & 15, half = lane >> 4;
    const int node = nb + m;
    const int qv = q[node], rv = rowi[node], cv = coli[node];
    u16* er = &sEmb[w][m * 64 + half * 32];
    for (int j = 0; j < 32; ++j) {
      const int k = half * 32 + j;
      u16 val;
      if (k < 16)       val = eD[qv * 16 + k];
      else if (k < 32)  val = eR[rv * 16 + (k - 16)];
      else if (k < 48)  val = eC[cv * 16 + (k - 32)];
      else              val = 0;
      er[j] = val;
    }
  }
  __syncthreads();

  const int m = lane & 15;
  const int mofs = (lane & 16) ? 8 : 0;
  // layer 0: K=64 (2 k-tiles)
  for (int nt = 0; nt < 6; ++nt) {
    v8f acc = {};
#pragma unroll
    for (int kt = 0; kt < 2; ++kt) {
      Frag a = load_a_bf(&sEmb[w][m * 64], kt * 32, lane);
      Frag b = load_b_bf(sW0p, 64, nt * 16 + (lane & 15), kt, lane);
      acc = wmma_bf16(a, b, acc);
    }
    const int n = nt * 16 + (lane & 15);
    const float bias = in_b[n];
#pragma unroll
    for (int r = 0; r < 8; ++r) {
      float v = acc[r] + bias; v = v > 0.f ? v : 0.f;
      sInterm[w][0][(r + mofs) * 96 + n] = f2bf(v);
    }
  }
  // layers 1..3 (K=96)
  int cur = 0;
  for (int L = 0; L < 3; ++L) {
    const u16* inb = &sInterm[w][cur][0];
    u16* outb = &sInterm[w][cur ^ 1][0];
    const u16* Wl = sW123 + L * 96 * 96;
    for (int nt = 0; nt < 6; ++nt) {
      v8f acc = {};
#pragma unroll
      for (int kt = 0; kt < 3; ++kt) {
        Frag a = load_a_bf(inb + m * 96, kt * 32, lane);
        Frag b = load_b_bf(Wl, 96, nt * 16 + (lane & 15), kt, lane);
        acc = wmma_bf16(a, b, acc);
      }
      const int n = nt * 16 + (lane & 15);
      const float bias = in_b[(L + 1) * 96 + n];
      if (L < 2) {
#pragma unroll
        for (int r = 0; r < 8; ++r) {
          float v = acc[r] + bias; v = v > 0.f ? v : 0.f;
          outb[(r + mofs) * 96 + n] = f2bf(v);
        }
      } else {
#pragma unroll
        for (int r = 0; r < 8; ++r) {
          const int node = nb + r + mofs;
          xbf[(size_t)node * HD + n] = f2bf(acc[r] + bias);  // no relu on final
        }
      }
    }
    cur ^= 1;
  }
}

// =====================================================================
// one recurrence step, fully fused per 16 destination nodes:
//   edge message MLP (320 edges, WMMA) -> LDS segment-sum -> LSTM gate
//   GEMMs (WMMA) -> elementwise LSTM -> logits GEMM (WMMA)
// =====================================================================
__global__ __launch_bounds__(256) void step_kernel(
    const u16* __restrict__ hbf_in, u16* __restrict__ hbf_out,
    const u16* __restrict__ xbf, float* __restrict__ rcb,
    const u16* __restrict__ wMsg0, const u16* __restrict__ wMsg,
    const u16* __restrict__ wIh, const u16* __restrict__ wHh,
    const u16* __restrict__ wOutP, const float* __restrict__ msg_b,
    const float* __restrict__ out_b, const int* __restrict__ dst,
    float* __restrict__ lf, int first) {
  __shared__ u16 sW0[SZ_MW0];          // msg layer0 weights (36KB)
  __shared__ u16 sW123[SZ_MW];         // msg layers 1-3 (54KB)
  __shared__ u16 sInterm[8][2][16 * 96];  // per-wave ping-pong activations (48KB)
  __shared__ float sM[16 * HD];        // aggregated messages, 16 dst nodes (6KB)
  __shared__ u16 sH[16 * HD];          // new h tile for logits (3KB)

  const int tid = threadIdx.x, w = tid >> 5, lane = tid & 31;
  const int v0 = blockIdx.x * 16;

  // stage hot weights in LDS; prefetch LSTM weights toward L2/L0
  {
    const uint4* s0 = (const uint4*)wMsg0; uint4* d0 = (uint4*)sW0;
    for (int i = tid; i < SZ_MW0 / 8; i += 256) d0[i] = s0[i];
    const uint4* s1 = (const uint4*)wMsg; uint4* d1 = (uint4*)sW123;
    for (int i = tid; i < SZ_MW / 8; i += 256) d1[i] = s1[i];
    if (tid < 64) __builtin_prefetch(wIh + tid * (SZ_WIH / 64), 0, 3);
    if (tid < 32) __builtin_prefetch(wHh + tid * (SZ_WHH / 32), 0, 3);
  }
  for (int i = tid; i < 16 * HD; i += 256) sM[i] = 0.f;
  __syncthreads();

  const int mofs = (lane & 16) ? 8 : 0;

  // ---------------- phase 1: edge messages, 20 tiles of 16 edges ----------------
  for (int te = w; te < DEG; te += 8) {
    const int le = te * 16 + (lane & 15);     // local edge id in [0,320)
    const int vl = le / DEG, ke = le % DEG;   // dst-local node, neighbor slot
    const int vnode = v0 + vl;
    const int unode = dst[vnode * DEG + ke];  // symmetric graph: incoming src
    const u16* rowU = hbf_in + (size_t)unode * HD;
    const u16* rowV = hbf_in + (size_t)vnode * HD;

    // layer 0: K = 192 = cat(h[u], h[v])
    u16* out0 = &sInterm[w][0][0];
    for (int nt = 0; nt < 6; ++nt) {
      v8f acc = {};
#pragma unroll
      for (int kt = 0; kt < 6; ++kt) {
        Frag a = load_a_bf(kt < 3 ? rowU : rowV, (kt % 3) * 32, lane);
        Frag b = load_b_bf(sW0, 192, nt * 16 + (lane & 15), kt, lane);
        acc = wmma_bf16(a, b, acc);
      }
      const int n = nt * 16 + (lane & 15);
      const float bias = msg_b[n];
#pragma unroll
      for (int r = 0; r < 8; ++r) {
        float v = acc[r] + bias; v = v > 0.f ? v : 0.f;
        out0[(r + mofs) * 96 + n] = f2bf(v);
      }
    }
    // layers 1..3 (K=96); final layer accumulates into sM (segment-sum)
    int cur = 0;
    for (int L = 0; L < 3; ++L) {
      const u16* inb = &sInterm[w][cur][0];
      u16* outb = &sInterm[w][cur ^ 1][0];
      const u16* Wl = sW123 + L * 96 * 96;
      for (int nt = 0; nt < 6; ++nt) {
        v8f acc = {};
#pragma unroll
        for (int kt = 0; kt < 3; ++kt) {
          Frag a = load_a_bf(inb + (lane & 15) * 96, kt * 32, lane);
          Frag b = load_b_bf(Wl, 96, nt * 16 + (lane & 15), kt, lane);
          acc = wmma_bf16(a, b, acc);
        }
        const int n = nt * 16 + (lane & 15);
        const float bias = msg_b[(L + 1) * 96 + n];
        if (L < 2) {
#pragma unroll
          for (int r = 0; r < 8; ++r) {
            float v = acc[r] + bias; v = v > 0.f ? v : 0.f;
            outb[(r + mofs) * 96 + n] = f2bf(v);
          }
        } else {
#pragma unroll
          for (int r = 0; r < 8; ++r) {
            const int mm = r + mofs;
            const int vl2 = (te * 16 + mm) / DEG;
            atomicAdd(&sM[vl2 * HD + n], acc[r] + bias);  // ds_add_f32
          }
        }
      }
      cur ^= 1;
    }
  }
  __syncthreads();

  // ---------------- phase 2: LSTM, wave s owns feature slice [16s,16s+16) ----------------
  if (w < 6) {
    const int n0 = w * 16;
    const int nloc = lane & 15;
    const int m = lane & 15;
    v8f gi = {}, gf = {}, gg = {}, go = {};
    const u16* xrow = xbf + (size_t)(v0 + m) * HD;
#pragma unroll
    for (int kt = 0; kt < 6; ++kt) {   // K=192: cat(x, m)
      Frag a = (kt < 3) ? load_a_bf(xrow, kt * 32, lane)
                        : load_a_f32(&sM[m * HD], (kt - 3) * 32, lane);
      gi = wmma_bf16(a, load_b_bf(wIh, 192,   0 + n0 + nloc, kt, lane), gi);
      gf = wmma_bf16(a, load_b_bf(wIh, 192,  96 + n0 + nloc, kt, lane), gf);
      gg = wmma_bf16(a, load_b_bf(wIh, 192, 192 + n0 + nloc, kt, lane), gg);
      go = wmma_bf16(a, load_b_bf(wIh, 192, 288 + n0 + nloc, kt, lane), go);
    }
    if (!first) {                      // rh == 0 on first step
      const u16* hrow = hbf_in + (size_t)(v0 + m) * HD;
#pragma unroll
      for (int kt = 0; kt < 3; ++kt) {
        Frag a = load_a_bf(hrow, kt * 32, lane);
        gi = wmma_bf16(a, load_b_bf(wHh, 96,   0 + n0 + nloc, kt, lane), gi);
        gf = wmma_bf16(a, load_b_bf(wHh, 96,  96 + n0 + nloc, kt, lane), gf);
        gg = wmma_bf16(a, load_b_bf(wHh, 96, 192 + n0 + nloc, kt, lane), gg);
        go = wmma_bf16(a, load_b_bf(wHh, 96, 288 + n0 + nloc, kt, lane), go);
      }
    }
#pragma unroll
    for (int r = 0; r < 8; ++r) {
      const int mm = r + mofs;
      const int node = v0 + mm;
      const int n = n0 + nloc;
      const size_t idx = (size_t)node * HD + n;
      const float iv = sigm(gi[r]);
      const float fv = sigm(gf[r]);
      const float gv = tanhf(gg[r]);
      const float ov = sigm(go[r]);
      const float nc = fv * rcb[idx] + iv * gv;
      const float nh = ov * tanhf(nc);
      rcb[idx] = nc;
      const u16 hb = f2bf(nh);
      hbf_out[idx] = hb;
      sH[mm * HD + n] = hb;
    }
  }
  __syncthreads();

  // ---------------- phase 3: logits (out 10 padded to 16), wave 0 ----------------
  if (w == 0) {
    const int m = lane & 15;
    v8f acc = {};
#pragma unroll
    for (int kt = 0; kt < 3; ++kt) {
      Frag a = load_a_bf(&sH[m * HD], kt * 32, lane);
      Frag b = load_b_bf(wOutP, 96, lane & 15, kt, lane);
      acc = wmma_bf16(a, b, acc);
    }
    const int n = lane & 15;
    if (n < 10) {
      const float bias = out_b[n];
#pragma unroll
      for (int r = 0; r < 8; ++r) {
        const int node = v0 + r + mofs;
        lf[(size_t)node * 10 + n] = acc[r] + bias;
      }
    }
  }
}

// =====================================================================
// preds (argmax) + NLL accumulation over T*N rows
// =====================================================================
__global__ __launch_bounds__(256) void loss_kernel(
    const float* __restrict__ lf, const int* __restrict__ labels,
    float* __restrict__ preds, float* __restrict__ acc) {
  __shared__ float red[256];
  const int i = blockIdx.x * 256 + threadIdx.x;
  float local = 0.f;
  if (i < TSTEPS * NN) {
    const float* l = lf + (size_t)i * 10;
    float mx = l[0]; int am = 0;
#pragma unroll
    for (int j = 1; j < 10; ++j) { const float v = l[j]; if (v > mx) { mx = v; am = j; } }
    float s = 0.f;
#pragma unroll
    for (int j = 0; j < 10; ++j) s += __expf(l[j] - mx);
    const float lse = mx + __logf(s);
    preds[i] = (float)am;
    const int lab = labels[i % NN];
    local = lse - l[lab];
  }
  red[threadIdx.x] = local;
  __syncthreads();
  for (int st = 128; st > 0; st >>= 1) {
    if (threadIdx.x < st) red[threadIdx.x] += red[threadIdx.x + st];
    __syncthreads();
  }
  if (threadIdx.x == 0) atomicAdd(acc, red[0]);
}

__global__ void finalize_kernel(float* lossOut, const float* acc) {
  *lossOut = acc[0] / (float)(TSTEPS * NN);
}

// =====================================================================
extern "C" void kernel_launch(void* const* d_in, const int* in_sizes, int n_in,
                              void* d_out, int out_size, void* d_ws, size_t ws_size,
                              hipStream_t stream) {
  const int*   q      = (const int*)d_in[0];
  const int*   rowi   = (const int*)d_in[1];
  const int*   coli   = (const int*)d_in[2];
  const int*   labels = (const int*)d_in[3];
  const int*   dst    = (const int*)d_in[5];
  const float* d_emb  = (const float*)d_in[6];
  const float* r_emb  = (const float*)d_in[7];
  const float* c_emb  = (const float*)d_in[8];
  const float* in_W0  = (const float*)d_in[9];
  const float* in_W   = (const float*)d_in[10];
  const float* in_b   = (const float*)d_in[11];
  const float* msg_W0 = (const float*)d_in[12];
  const float* msg_W  = (const float*)d_in[13];
  const float* msg_b  = (const float*)d_in[14];
  const float* W_ih   = (const float*)d_in[15];
  const float* W_hh   = (const float*)d_in[16];
  const float* out_W  = (const float*)d_in[17];
  const float* out_b  = (const float*)d_in[18];

  // workspace carve-up (bf16 regions first, then f32)
  u16* p = (u16*)d_ws;
  u16* wInW0p = p; p += SZ_INW0P;
  u16* wInW   = p; p += SZ_INW;
  u16* wMW0   = p; p += SZ_MW0;
  u16* wMW    = p; p += SZ_MW;
  u16* wIh    = p; p += SZ_WIH;
  u16* wHh    = p; p += SZ_WHH;
  u16* wOutP  = p; p += SZ_OUTWP;
  u16* eD     = p; p += SZ_DE;
  u16* eR     = p; p += SZ_RE;
  u16* eC     = p; p += SZ_CE;
  u16* xbf    = p; p += SZ_ACT;
  u16* hA     = p; p += SZ_ACT;
  u16* hB     = p; p += SZ_ACT;
  float* rcb     = (float*)p;
  float* lossAcc = rcb + SZ_ACT;

  float* out = (float*)d_out;
  const size_t LFOFF = (size_t)TSTEPS * NN + 1;

  prep_kernel<<<(PREP_TOTAL + 255) / 256, 256, 0, stream>>>(
      in_W0, in_W, msg_W0, msg_W, W_ih, W_hh, out_W, d_emb, r_emb, c_emb,
      wInW0p, wInW, wMW0, wMW, wIh, wHh, wOutP, eD, eR, eC, rcb, lossAcc);

  input_kernel<<<NN / 128, 256, 0, stream>>>(q, rowi, coli, eD, eR, eC, wInW0p,
                                             wInW, in_b, xbf);

  for (int t = 0; t < TSTEPS; ++t) {
    const u16* hin = (t == 0) ? xbf : ((t & 1) ? hA : hB);
    u16* hout = (t & 1) ? hB : hA;
    float* lf_t = out + LFOFF + (size_t)t * NN * 10;
    step_kernel<<<NN / 16, 256, 0, stream>>>(hin, hout, xbf, rcb, wMW0, wMW, wIh,
                                             wHh, wOutP, msg_b, out_b, dst, lf_t,
                                             t == 0 ? 1 : 0);
  }

  loss_kernel<<<(TSTEPS * NN) / 256, 256, 0, stream>>>(out + LFOFF, labels, out,
                                                       lossAcc);
  finalize_kernel<<<1, 1, 0, stream>>>(out + (size_t)TSTEPS * NN, lossAcc);
}
